// GraphConvolution_75728863363703
// MI455X (gfx1250) — compile-verified, matrix operand order checked by reading, not run on hardware
//
#include <hip/hip_runtime.h>
#include <hip/hip_bf16.h>
#include <math.h>

typedef __attribute__((ext_vector_type(16))) __bf16 v16bf;
typedef __attribute__((ext_vector_type(8)))  float  v8f;

#define N_NODES 8192
#define HDIM    256
#define KC      32      // K per chunk (one bf16 WMMA step)
#define MT      32      // rows per workgroup
#define LDA     48      // padded LDS row stride in bf16 elems (96B; 16B-aligned rows)

union Frag { uint4 u[2]; v16bf v; };

// ---------------------------------------------------------------------------
// Kernel 1: pre_sup = inputs @ W  (fp32 accurate), also emit bf16 transpose
// preT[H][N] so the big matmul's B operand is K-contiguous per column.
// ---------------------------------------------------------------------------
__global__ __launch_bounds__(256)
void presup_kernel(const float* __restrict__ inp, const float* __restrict__ W,
                   float* __restrict__ pre, __bf16* __restrict__ preT)
{
    __shared__ float sIn[4][HDIM];
    const int r0 = blockIdx.x * 4;
    const int c  = threadIdx.x;          // 0..255 = output column

    #pragma unroll
    for (int rr = 0; rr < 4; ++rr)
        sIn[rr][c] = inp[(size_t)(r0 + rr) * HDIM + c];
    __syncthreads();

    float s0 = 0.f, s1 = 0.f, s2 = 0.f, s3 = 0.f;
    for (int k = 0; k < HDIM; ++k) {
        float w = W[(size_t)k * HDIM + c];   // coalesced; W is L2-resident
        s0 = fmaf(sIn[0][k], w, s0);
        s1 = fmaf(sIn[1][k], w, s1);
        s2 = fmaf(sIn[2][k], w, s2);
        s3 = fmaf(sIn[3][k], w, s3);
    }
    pre[(size_t)(r0 + 0) * HDIM + c] = s0;
    pre[(size_t)(r0 + 1) * HDIM + c] = s1;
    pre[(size_t)(r0 + 2) * HDIM + c] = s2;
    pre[(size_t)(r0 + 3) * HDIM + c] = s3;
    preT[(size_t)c * N_NODES + r0 + 0] = (__bf16)s0;
    preT[(size_t)c * N_NODES + r0 + 1] = (__bf16)s1;
    preT[(size_t)c * N_NODES + r0 + 2] = (__bf16)s2;
    preT[(size_t)c * N_NODES + r0 + 3] = (__bf16)s3;
}

// ---------------------------------------------------------------------------
// Kernel 2: single pass over adj. Fused: rowsum(deg) + WMMA matmul +
// self-loop correction + 1/deg scale + bias + tanh.
//
// Inner loop: manually unrolled x2 (compile-time ping-pong buffer index),
// single running pointer per stream advanced once per iteration; all
// fragment/prefetch loads use compile-time immediate offsets (fit in the
// 24-bit signed IOFFSET), so per-iteration address VALU is ~2 u64 adds.
// ---------------------------------------------------------------------------
__global__ __launch_bounds__(256)
void gcn_main_kernel(const float* __restrict__ adj, const __bf16* __restrict__ preT,
                     const float* __restrict__ pre, const float* __restrict__ bias,
                     float* __restrict__ out)
{
    __shared__ __bf16 sA[2][MT * LDA];   // double-buffered bf16 adj tile (6KB)
    __shared__ float  sR[256];           // rowsum staging
    __shared__ float  sDeg[MT];          // reciprocal degree

    const int tid  = threadIdx.x;
    const int wave = tid >> 5;
    const int lane = tid & 31;
    const int r0   = blockIdx.x * MT;

    const int rt    = wave & 1;          // row tile (0/1)
    const int ct0   = (wave >> 1) * 4;   // first of 4 col tiles for this wave
    const int lhalf = lane >> 4;         // lane group (K swizzle select)
    const int lmod  = lane & 15;

    // cooperative A-tile loader assignment: one float4 of adj per thread/chunk
    const int lrow = tid >> 3;           // 0..31
    const int lkq  = tid & 7;            // 0..7 (float4 quad within chunk)

    // running pointers (advanced by 2*KC per loop iteration)
    const float*  aptr  = adj + (size_t)(r0 + lrow) * N_NODES + lkq * 4;
    const __bf16* bptr  = preT + (size_t)(ct0 * 16 + lmod) * N_NODES + lhalf * 8;
    // fixed LDS addresses (buffer selected via immediate offset)
    const __bf16* abase = &sA[0][(rt * 16 + lmod) * LDA + lhalf * 8];
    __bf16*       sdst  = &sA[0][lrow * LDA + lkq * 4];

    float rsum = 0.f;
    v8f acc[4];
    #pragma unroll
    for (int t = 0; t < 4; ++t)
        #pragma unroll
        for (int j = 0; j < 8; ++j) acc[t][j] = 0.f;

    // preload chunk 0 into buffer 0
    {
        float4 a4 = *(const float4*)(aptr);
        rsum += (a4.x + a4.y) + (a4.z + a4.w);
        sdst[0] = (__bf16)a4.x; sdst[1] = (__bf16)a4.y;
        sdst[2] = (__bf16)a4.z; sdst[3] = (__bf16)a4.w;
    }
    __syncthreads();

    const int NCHUNK = N_NODES / KC;     // 256 (even)
    #pragma clang loop unroll_count(1)
    for (int kc = 0; kc < NCHUNK; kc += 2) {
        // ---------- half 0: compute buf0 (chunk kc), prefetch kc+1 -> buf1
        {
            float4 a4 = *(const float4*)(aptr + KC);        // imm offset 128B

            Frag af;
            af.u[0] = *(const uint4*)(abase);               // ds offset 0
            af.u[1] = *(const uint4*)(abase + 16);          // ds offset 32B

            #pragma unroll
            for (int t = 0; t < 4; ++t) {
                const __bf16* bb = bptr + (size_t)t * 16 * N_NODES;  // imm t*256KB
                Frag bf_;
                bf_.u[0] = *(const uint4*)(bb);
                bf_.u[1] = *(const uint4*)(bb + 16);
                acc[t] = __builtin_amdgcn_wmma_f32_16x16x32_bf16(
                    false, af.v, false, bf_.v, (short)0, acc[t], false, false);
            }

            rsum += (a4.x + a4.y) + (a4.z + a4.w);
            __bf16* dst = sdst + MT * LDA;                  // buf1, imm offset
            dst[0] = (__bf16)a4.x; dst[1] = (__bf16)a4.y;
            dst[2] = (__bf16)a4.z; dst[3] = (__bf16)a4.w;
            __syncthreads();
        }

        // ---------- half 1: compute buf1 (chunk kc+1), prefetch kc+2 -> buf0
        {
            const bool more = (kc + 2) < NCHUNK;
            float4 a4;
            if (more) a4 = *(const float4*)(aptr + 2 * KC); // imm offset 256B

            Frag af;
            af.u[0] = *(const uint4*)(abase + MT * LDA);    // buf1 ds offsets
            af.u[1] = *(const uint4*)(abase + MT * LDA + 16);

            #pragma unroll
            for (int t = 0; t < 4; ++t) {
                const __bf16* bb = bptr + (size_t)t * 16 * N_NODES + KC;
                Frag bf_;
                bf_.u[0] = *(const uint4*)(bb);
                bf_.u[1] = *(const uint4*)(bb + 16);
                acc[t] = __builtin_amdgcn_wmma_f32_16x16x32_bf16(
                    false, af.v, false, bf_.v, (short)0, acc[t], false, false);
            }

            if (more) {
                rsum += (a4.x + a4.y) + (a4.z + a4.w);
                sdst[0] = (__bf16)a4.x; sdst[1] = (__bf16)a4.y;
                sdst[2] = (__bf16)a4.z; sdst[3] = (__bf16)a4.w;
            }
            __syncthreads();
        }

        aptr += 2 * KC;                  // one u64 add per stream per iteration
        bptr += 2 * KC;
    }

    // degree reduction: deg[r] = 1 (self loop) + sum over 8 partial sums
    sR[tid] = rsum;
    __syncthreads();
    if (tid < MT) {
        float d = 1.0f;
        #pragma unroll
        for (int i = 0; i < 8; ++i) d += sR[tid * 8 + i];
        sDeg[tid] = 1.0f / d;
    }
    __syncthreads();

    // epilogue: diag correction (fp32), normalize, bias, tanh
    #pragma unroll
    for (int t = 0; t < 4; ++t) {
        const int c  = (ct0 + t) * 16 + lmod;
        const float bc = bias[c];
        #pragma unroll
        for (int j = 0; j < 8; ++j) {
            const int m   = j + 8 * lhalf;
            const int row = rt * 16 + m;
            const int r   = r0 + row;
            float v = (acc[t][j] + pre[(size_t)r * HDIM + c]) * sDeg[row] + bc;
            out[(size_t)r * HDIM + c] = tanhf(v);
        }
    }
}

// ---------------------------------------------------------------------------
extern "C" void kernel_launch(void* const* d_in, const int* in_sizes, int n_in,
                              void* d_out, int out_size, void* d_ws, size_t ws_size,
                              hipStream_t stream) {
    const float* inp = (const float*)d_in[0];   // [8192,256]
    const float* adj = (const float*)d_in[1];   // [8192,8192]
    const float* W   = (const float*)d_in[2];   // [256,256]
    const float* b   = (const float*)d_in[3];   // [256]
    float* out = (float*)d_out;                 // [8192,256]

    // workspace: preT bf16 [H][N] (4MB) then pre fp32 [N][H] (8MB)
    __bf16* preT = (__bf16*)d_ws;
    float*  pre  = (float*)((char*)d_ws + (size_t)HDIM * N_NODES * sizeof(__bf16));

    presup_kernel<<<N_NODES / 4, 256, 0, stream>>>(inp, W, pre, preT);
    gcn_main_kernel<<<N_NODES / MT, 256, 0, stream>>>(adj, preT, pre, b, out);
}
